// RGCN_Encoder_49778670961012
// MI455X (gfx1250) — compile-verified, matrix operand order checked by reading, not run on hardware
//
#include <hip/hip_runtime.h>
#include <hip/hip_bf16.h>

// ---------------------------------------------------------------------------
// RGCN 2-layer encoder for MI455X (gfx1250, wave32, WMMA).
// Strategy:
//  * bf16 WMMA (v_wmma_f32_16x16x32_bf16) for all dense transforms; root
//    weight folded in as "relation 8" of an augmented weight stack.
//  * mean-over-(dst,rel) then sum-over-rel collapses (linearity) into ONE
//    scatter pass: out_agg[dst] += h[src,et] * (1/max(cnt[dst,et],1)).
//  * HBM-bound by design: h streams (230MB), out_agg (25.6MB) lives in L2
//    during the f32 atomic scatter.
// ---------------------------------------------------------------------------

#define NNODES 50000
#define NEDGES 800000
#define HDIM   128
#define NREL   8
#define NMAT   (NREL + 1)   // +1 for the root weight

typedef __attribute__((ext_vector_type(16))) __bf16 v16bf;
typedef __attribute__((ext_vector_type(8)))  __bf16 v8bf;
typedef __attribute__((ext_vector_type(8)))  float  v8f;

static inline int ceil_div_i(long long a, long long b) { return (int)((a + b - 1) / b); }

// ------------------------------ utility kernels ----------------------------

__global__ __launch_bounds__(256) void k_zero_f32(float* p, long long n) {
  long long i = (long long)blockIdx.x * blockDim.x + threadIdx.x;
  if (i < n) p[i] = 0.0f;
}

__global__ __launch_bounds__(256) void k_count_edges(const int* __restrict__ dst,
                                                     const int* __restrict__ et,
                                                     float* __restrict__ cnt) {
  long long e = (long long)blockIdx.x * blockDim.x + threadIdx.x;
  if (e >= NEDGES) return;
  atomicAdd(&cnt[(long long)dst[e] * NREL + et[e]], 1.0f);
}

__global__ __launch_bounds__(256) void k_recip(const float* __restrict__ cnt,
                                               float* __restrict__ inv, long long n) {
  long long i = (long long)blockIdx.x * blockDim.x + threadIdx.x;
  if (i < n) inv[i] = 1.0f / fmaxf(cnt[i], 1.0f);
}

// Build bf16 input for layer 1: xb[n][0:128]=x0[n], xb[n][128:256]=x1[n]
__global__ __launch_bounds__(256) void k_build_x1(const float* __restrict__ x0,
                                                  const float* __restrict__ x1,
                                                  __hip_bfloat16* __restrict__ xb) {
  long long i = (long long)blockIdx.x * blockDim.x + threadIdx.x;
  if (i >= (long long)NNODES * 256) return;
  int k = (int)(i & 255);
  long long n = i >> 8;
  float v = (k < HDIM) ? x0[n * HDIM + k] : x1[n * HDIM + (k - HDIM)];
  xb[i] = __float2bfloat16(v);
}

// Wt[mat][o][k] (transposed, bf16) from W[r][k][o] (f32) and root[k][o]
__global__ __launch_bounds__(256) void k_build_wt(const float* __restrict__ W,
                                                  const float* __restrict__ root,
                                                  __hip_bfloat16* __restrict__ Wt,
                                                  int Din) {
  long long i = (long long)blockIdx.x * blockDim.x + threadIdx.x;
  long long total = (long long)NMAT * HDIM * Din;
  if (i >= total) return;
  int k   = (int)(i % Din);
  int o   = (int)((i / Din) % HDIM);
  int mat = (int)(i / ((long long)Din * HDIM));
  float v = (mat < NREL) ? W[((long long)mat * Din + k) * HDIM + o]
                         : root[(long long)k * HDIM + o];
  Wt[i] = __float2bfloat16(v);
}

// ------------------------------ WMMA batched GEMM --------------------------
// Hout[n][mat][o] = A[n][:] . Wt[mat][o][:]   for mat in 0..8
// One wave computes one 16(M)x16(N) f32 tile; 8 waves/block share an A slab.
// tile id = rowt*72 + mat*8 + colt  (3125 row tiles, 9 mats, 8 col tiles)
__global__ __launch_bounds__(256)
void k_rgcn_gemm_bf16(const __hip_bfloat16* __restrict__ A,   // [NNODES, Din]
                      const __hip_bfloat16* __restrict__ Wt,  // [NMAT, HDIM, Din]
                      float* __restrict__ Hout,               // [NNODES, NMAT, HDIM]
                      int Din) {
  const int lane = threadIdx.x & 31;
  const int wave = threadIdx.x >> 5;
  const int tile = blockIdx.x * 8 + wave;
  const int colt = tile & 7;
  const int mat  = (tile >> 3) % NMAT;
  const int rowt = tile / (8 * NMAT);
  const int m0 = rowt * 16;
  const int n0 = colt * 16;

  const int half = lane >> 4;   // 0: lanes 0-15, 1: lanes 16-31
  const int lidx = lane & 15;
  const int kb   = half * 8;    // 16-bit A/B fragment K sub-offset per half

  const __hip_bfloat16* arow = A  + (long long)(m0 + lidx) * Din;
  const __hip_bfloat16* bcol = Wt + ((long long)mat * HDIM + (n0 + lidx)) * Din;

  v8f acc = {};
  for (int kk = 0; kk < Din; kk += 32) {
    union { v16bf v; v8bf h[2]; } a, b;
    // A fragment: VGPR0-3 <- K = kb..kb+7 ; VGPR4-7 <- K = 16+kb..16+kb+7
    a.h[0] = *reinterpret_cast<const v8bf*>(arow + kk + kb);
    a.h[1] = *reinterpret_cast<const v8bf*>(arow + kk + 16 + kb);
    // B fragment mirrors A with N-column in place of M-row (Wt is transposed)
    b.h[0] = *reinterpret_cast<const v8bf*>(bcol + kk + kb);
    b.h[1] = *reinterpret_cast<const v8bf*>(bcol + kk + 16 + kb);
    acc = __builtin_amdgcn_wmma_f32_16x16x32_bf16(
        /*neg_a=*/false, a.v, /*neg_b=*/false, b.v,
        /*c_mod=*/(short)0, acc, /*reuse_a=*/false, /*reuse_b=*/false);
  }

  // D layout: VGPR v holds (M = m0 + v + 8*half, N = n0 + lidx)
  #pragma unroll
  for (int v = 0; v < 8; ++v) {
    long long row = m0 + v + 8 * half;
    Hout[(row * NMAT + mat) * HDIM + n0 + lidx] = acc[v];
  }
}

// ------------------------------ aggregation --------------------------------

// out_agg[n][j] = h[n][8][j] (root term) + b[j]
__global__ __launch_bounds__(256) void k_init_out(const float* __restrict__ Hh,
                                                  const float* __restrict__ bias,
                                                  float* __restrict__ acc) {
  long long i = (long long)blockIdx.x * blockDim.x + threadIdx.x;
  if (i >= (long long)NNODES * HDIM) return;
  int j = (int)(i & (HDIM - 1));
  long long n = i >> 7;
  acc[i] = Hh[(n * NMAT + NREL) * HDIM + j] + bias[j];
}

// One wave per edge: acc[dst] += h[src,et] * inv[dst*NREL+et]  (float4 loads)
__global__ __launch_bounds__(256)
void k_scatter(const float* __restrict__ Hh, const int* __restrict__ src,
               const int* __restrict__ dst, const int* __restrict__ et,
               const float* __restrict__ inv, float* __restrict__ acc) {
  long long t = (long long)blockIdx.x * blockDim.x + threadIdx.x;
  long long e = t >> 5;
  if (e >= NEDGES) return;
  int lane = (int)(t & 31);
  int s = src[e], d = dst[e], r = et[e];
  float sc = inv[(long long)d * NREL + r];
  const float4 v = reinterpret_cast<const float4*>(
      Hh + ((long long)s * NMAT + r) * HDIM)[lane];
  float* out = acc + (long long)d * HDIM + lane * 4;
  atomicAdd(out + 0, v.x * sc);
  atomicAdd(out + 1, v.y * sc);
  atomicAdd(out + 2, v.z * sc);
  atomicAdd(out + 3, v.w * sc);
}

__global__ __launch_bounds__(256) void k_relu_to_bf16(const float* __restrict__ acc,
                                                      __hip_bfloat16* __restrict__ xb) {
  long long i = (long long)blockIdx.x * blockDim.x + threadIdx.x;
  if (i >= (long long)NNODES * HDIM) return;
  xb[i] = __float2bfloat16(fmaxf(acc[i], 0.0f));
}

__global__ __launch_bounds__(256) void k_relu_to_f32(const float* __restrict__ acc,
                                                     float* __restrict__ out) {
  long long i = (long long)blockIdx.x * blockDim.x + threadIdx.x;
  if (i >= (long long)NNODES * HDIM) return;
  out[i] = fmaxf(acc[i], 0.0f);
}

// ------------------------------ launcher -----------------------------------

extern "C" void kernel_launch(void* const* d_in, const int* in_sizes, int n_in,
                              void* d_out, int out_size, void* d_ws, size_t ws_size,
                              hipStream_t stream) {
  (void)in_sizes; (void)n_in; (void)out_size; (void)ws_size;

  const float* x0    = (const float*)d_in[0];
  const float* x1    = (const float*)d_in[1];
  const int*   eidx  = (const int*)d_in[2];       // [2, E] flat
  const int*   etype = (const int*)d_in[3];
  const float* W1    = (const float*)d_in[4];     // [R, 256, 128]
  const float* root1 = (const float*)d_in[5];     // [256, 128]
  const float* b1    = (const float*)d_in[6];
  const float* W2    = (const float*)d_in[7];     // [R, 128, 128]
  const float* root2 = (const float*)d_in[8];
  const float* b2    = (const float*)d_in[9];
  const int* src = eidx;
  const int* dst = eidx + NEDGES;
  float* out = (float*)d_out;

  // ---- workspace carve-up (256B aligned), total ~298 MB ----
  char* base = (char*)d_ws;
  size_t off = 0;
  auto carve = [&](size_t bytes) -> void* {
    void* p = base + off;
    off = (off + bytes + 255) & ~(size_t)255;
    return p;
  };
  __hip_bfloat16* xb1 = (__hip_bfloat16*)carve((size_t)NNODES * 256 * 2);   // 25.6 MB
  __hip_bfloat16* xb2 = (__hip_bfloat16*)carve((size_t)NNODES * HDIM * 2);  // 12.8 MB
  __hip_bfloat16* Wt1 = (__hip_bfloat16*)carve((size_t)NMAT * HDIM * 256 * 2);
  __hip_bfloat16* Wt2 = (__hip_bfloat16*)carve((size_t)NMAT * HDIM * HDIM * 2);
  float* Hbuf = (float*)carve((size_t)NNODES * NMAT * HDIM * 4);            // 230.4 MB
  float* aggb = (float*)carve((size_t)NNODES * HDIM * 4);                   // 25.6 MB
  float* cnt  = (float*)carve((size_t)NNODES * NREL * 4);
  float* inv  = (float*)carve((size_t)NNODES * NREL * 4);

  const int T = 256;
  const long long nNR   = (long long)NNODES * NREL;
  const long long nNH   = (long long)NNODES * HDIM;
  const int gemm_blocks = (3125 * 8 * NMAT) / 8;   // 28125: one 16x16 tile per wave

  // ---- graph-invariant degree factors ----
  k_zero_f32<<<ceil_div_i(nNR, T), T, 0, stream>>>(cnt, nNR);
  k_count_edges<<<ceil_div_i(NEDGES, T), T, 0, stream>>>(dst, etype, cnt);
  k_recip<<<ceil_div_i(nNR, T), T, 0, stream>>>(cnt, inv, nNR);

  // ---- bf16 operand prep ----
  k_build_x1<<<ceil_div_i((long long)NNODES * 256, T), T, 0, stream>>>(x0, x1, xb1);
  k_build_wt<<<ceil_div_i((long long)NMAT * HDIM * 256, T), T, 0, stream>>>(W1, root1, Wt1, 256);
  k_build_wt<<<ceil_div_i((long long)NMAT * HDIM * HDIM, T), T, 0, stream>>>(W2, root2, Wt2, HDIM);

  // ---- layer 1 ----
  k_rgcn_gemm_bf16<<<gemm_blocks, T, 0, stream>>>(xb1, Wt1, Hbuf, 256);
  k_init_out<<<ceil_div_i(nNH, T), T, 0, stream>>>(Hbuf, b1, aggb);
  k_scatter<<<ceil_div_i((long long)NEDGES * 32, T), T, 0, stream>>>(Hbuf, src, dst, etype, inv, aggb);
  k_relu_to_bf16<<<ceil_div_i(nNH, T), T, 0, stream>>>(aggb, xb2);

  // ---- layer 2 ----
  k_rgcn_gemm_bf16<<<gemm_blocks, T, 0, stream>>>(xb2, Wt2, Hbuf, HDIM);
  k_init_out<<<ceil_div_i(nNH, T), T, 0, stream>>>(Hbuf, b2, aggb);
  k_scatter<<<ceil_div_i((long long)NEDGES * 32, T), T, 0, stream>>>(Hbuf, src, dst, etype, inv, aggb);
  k_relu_to_f32<<<ceil_div_i(nNH, T), T, 0, stream>>>(aggb, out);
}
